// SpaM_13984413516036
// MI455X (gfx1250) — matrix operand order repaired and checked by reading, not compiled, not run on hardware
//
#include <hip/hip_runtime.h>
#include <math.h>

// ---------------------------------------------------------------------------
// MI455X (gfx1250) signed-GNN forward, v2.
//  - All GEMMs via v_wmma_f32_16x16x32_bf16, wave32, 16x32 output per wave
//    (two WMMAs share one A fragment; NT=1 variant for the 16-wide classifier).
//  - Every GEMM A-operand is kept as a bf16 shadow buffer written by its
//    producer (GEMM epilogue / elementwise epilogue / convert pass), so the
//    WMMA inner loop does NO float->bf16 conversion: A fragment = two 16B
//    loads + shufflevector, B fragment = one 32B load (pre-transposed bf16).
//  - Edge gathers (T[col], V[row], hs[row/col]) fused into A addressing.
//  - Segment-sums via global fp32 atomics (relaxed, agent scope).
// ---------------------------------------------------------------------------

typedef __attribute__((ext_vector_type(8)))  __bf16 v8bf;
typedef __attribute__((ext_vector_type(16))) __bf16 v16bf;
typedef __attribute__((ext_vector_type(8)))  float  v8f;

#define N_NODES 20000
#define N_EDGES 160000
#define DIM_IN  256
#define DIM_HID 128
#define DIM_VAL 128
#define N_CLS   16
#define K_MC    3
#define DIM_SEMB 8
#define LAM_F 0.1f
#define GAMMA_F 1.0f

static_assert(N_NODES % 16 == 0 && N_EDGES % 16 == 0, "M tiles exact");
static_assert(DIM_IN % 32 == 0 && DIM_HID % 32 == 0, "K tiles exact");

__device__ __forceinline__ unsigned short f2bf_bits(float f) {
  unsigned u = __builtin_bit_cast(unsigned, f);
  u += 0x7FFFu + ((u >> 16) & 1u);          // round-to-nearest-even
  return (unsigned short)(u >> 16);
}
__device__ __forceinline__ __bf16 f2bf(float f) {
  unsigned short s = f2bf_bits(f);
  return __builtin_bit_cast(__bf16, s);
}
__device__ __forceinline__ void atomAddF(float* p, float v) {
  __hip_atomic_fetch_add(p, v, __ATOMIC_RELAXED, __HIP_MEMORY_SCOPE_AGENT);
}

// ---------------------------------------------------------------------------
// WMMA GEMM: C[M,Nd] (+=) A[gidx?,Kd] @ W[Kd,Nd]; A bf16, W as bf16 W^T[Nd,Kd].
// grid = (M/16, Nd/(16*NT)), block = 32 (one wave), NT tiles of 16 cols/wave.
// flags: 1=acc from C, 2=+bias[n], 4=relu, 8=+resid, 16=also store bf16 copy.
// ---------------------------------------------------------------------------
template <int NT>
__global__ __launch_bounds__(32)
void gemm_wmma(const __bf16* __restrict__ A, const int* __restrict__ gidx,
               const __bf16* __restrict__ BT,
               const float* __restrict__ bias, const float* __restrict__ resid,
               float* __restrict__ Cm, __bf16* __restrict__ Cbf,
               int Kdim, int Ndim, int flags)
{
  const int lane = threadIdx.x;
  const int hi   = lane >> 4;               // half-wave select
  const int l15  = lane & 15;
  const int m    = blockIdx.x * 16 + l15;   // A-fragment row for this lane
  const int n0   = blockIdx.y * (16 * NT);

  const int arow = gidx ? gidx[m] : m;
  const __bf16* aptr = A + (long long)arow * Kdim + hi * 16;
  const int mb = blockIdx.x * 16 + hi * 8;  // C-fragment base row

  v8f acc[NT];
  if (flags & 1) {
#pragma unroll
    for (int t = 0; t < NT; ++t)
#pragma unroll
      for (int v = 0; v < 8; ++v)
        acc[t][v] = Cm[(long long)(mb + v) * Ndim + n0 + t * 16 + l15];
  } else {
#pragma unroll
    for (int t = 0; t < NT; ++t)
#pragma unroll
      for (int v = 0; v < 8; ++v) acc[t][v] = 0.0f;
  }

  for (int kk = 0; kk < Kdim; kk += 32) {
    // A fragment: k = hi*16 + e; two 16B loads
    const v8bf alo = *(const v8bf*)(aptr + kk);
    const v8bf ahi = *(const v8bf*)(aptr + kk + 8);
    const v16bf af = __builtin_shufflevector(alo, ahi, 0, 1, 2, 3, 4, 5, 6, 7,
                                             8, 9, 10, 11, 12, 13, 14, 15);
#pragma unroll
    for (int t = 0; t < NT; ++t) {
      // B fragment: lane n fixed, k = hi*16 + e -> one 32B load
      const v16bf bv = *(const v16bf*)(BT + (long long)(n0 + t * 16 + l15) * Kdim
                                       + hi * 16 + kk);
      acc[t] = __builtin_amdgcn_wmma_f32_16x16x32_bf16(false, af, false, bv,
                                                       (short)0, acc[t], false, false);
    }
  }

#pragma unroll
  for (int t = 0; t < NT; ++t) {
    const int n = n0 + t * 16 + l15;
    const float bn = (flags & 2) ? bias[n] : 0.0f;
#pragma unroll
    for (int v = 0; v < 8; ++v) {
      const long long off = (long long)(mb + v) * Ndim + n;
      float r = acc[t][v] + bn;
      if (flags & 8)  r += resid[off];
      if (flags & 4)  r = fmaxf(r, 0.0f);
      Cm[off] = r;
      if (flags & 16) Cbf[off] = f2bf(r);
    }
  }
}

// ----------------------------- small kernels -------------------------------
__global__ void k_fill(float* p, int n, float v) {
  int i = blockIdx.x * blockDim.x + threadIdx.x;
  if (i < n) p[i] = v;
}
__global__ void k_f2bf16(const float* __restrict__ in, __bf16* __restrict__ out, int n) {
  int i = blockIdx.x * blockDim.x + threadIdx.x;
  if (i < n) out[i] = f2bf(in[i]);
}
__global__ void k_deg(const int* __restrict__ ei, float* __restrict__ deg) {
  int e = blockIdx.x * blockDim.x + threadIdx.x;
  if (e < N_EDGES) atomAddF(&deg[ei[N_EDGES + e]], 1.0f);
}
__global__ void k_rsqrt(float* p, int n) {
  int i = blockIdx.x * blockDim.x + threadIdx.x;
  if (i < n) p[i] = rsqrtf(p[i]);
}
// self-loop term of GCN aggregation: out = dinv_i^2 * h
__global__ void k_selfscale(const float* __restrict__ h, const float* __restrict__ dinv,
                            float* __restrict__ out) {
  int t = blockIdx.x * blockDim.x + threadIdx.x;
  if (t >= N_NODES * DIM_HID) return;
  float d = dinv[t >> 7];
  out[t] = d * d * h[t];
}
// out[col] += dinv[row]*dinv[col] * h[row], 4 feats/thread
__global__ void k_gcn_edge(const float* __restrict__ h, const int* __restrict__ ei,
                           const float* __restrict__ dinv, float* __restrict__ out) {
  int t = blockIdx.x * blockDim.x + threadIdx.x;
  if (t >= N_EDGES * 32) return;
  int e = t >> 5, f = (t & 31) << 2;
  int r = ei[e], c = ei[N_EDGES + e];
  float nrm = dinv[r] * dinv[c];
  const float4 hv = *(const float4*)(h + r * DIM_HID + f);
  float* o = out + c * DIM_HID + f;
  atomAddF(o + 0, nrm * hv.x); atomAddF(o + 1, nrm * hv.y);
  atomAddF(o + 2, nrm * hv.z); atomAddF(o + 3, nrm * hv.w);
}
// out = [relu]( in [+bias[f]] [+resid] ); optional bf16 shadow store (flag 8)
__global__ void k_epilogue(const float* __restrict__ in, const float* __restrict__ bias,
                           const float* __restrict__ resid, float* __restrict__ out,
                           __bf16* __restrict__ outb, int total, int flags) {
  int t = blockIdx.x * blockDim.x + threadIdx.x;
  if (t >= total) return;
  float r = in[t];
  if (flags & 1) r += bias[t & (DIM_HID - 1)];
  if (flags & 2) r += resid[t];
  if (flags & 4) r = fmaxf(r, 0.0f);
  out[t] = r;
  if (flags & 8) outb[t] = f2bf(r);
}
__global__ void k_edge_logits(const float* __restrict__ ef, const float* __restrict__ w,
                              const float* __restrict__ b, float* __restrict__ elog) {
  int e = blockIdx.x * blockDim.x + threadIdx.x;
  if (e >= N_EDGES) return;
  const float* hr = ef + (long long)e * DIM_HID;
  float l0 = b[0], l1 = b[1], l2 = b[2];
  for (int k2 = 0; k2 < DIM_HID; ++k2) {
    float h = hr[k2];
    l0 += h * w[k2 * 3 + 0]; l1 += h * w[k2 * 3 + 1]; l2 += h * w[k2 * 3 + 2];
  }
  elog[e * 3 + 0] = l0; elog[e * 3 + 1] = l1; elog[e * 3 + 2] = l2;
}
__global__ void k_struct_loss(const float* __restrict__ elog, float* __restrict__ scal) {
  int e = blockIdx.x * blockDim.x + threadIdx.x;
  if (e >= N_EDGES) return;
  float l0 = elog[e * 3], l1 = elog[e * 3 + 1], l2 = elog[e * 3 + 2];
  float m = fmaxf(l0, fmaxf(l1, l2));
  float e0 = expf(l0 - m), e1 = expf(l1 - m), e2 = expf(l2 - m);
  float s = e0 + e1 + e2, lse = m + logf(s);
  float p0 = e0 / s, p1 = e1 / s, p2 = e2 / s;
  const float LP = -1.09861228866810969f;   // log(1/3 + 1e-12)
  float kl = p0 * (l0 - lse - LP) + p1 * (l1 - lse - LP) + p2 * (l2 - lse - LP);
  atomAddF(&scal[1], kl);
  atomAddF(&scal[2], logf(p0 + p2 + 1e-12f));
}
// forward value of straight-through gumbel-softmax == one_hot(argmax(logits+g))
__global__ void k_sign(const float* __restrict__ elog, const float* __restrict__ gu,
                       int* __restrict__ sidx) {
  int e = blockIdx.x * blockDim.x + threadIdx.x;
  if (e >= N_EDGES) return;
  float best = -1e30f; int bi = 0;
#pragma unroll
  for (int j = 0; j < 3; ++j) {
    float z = elog[e * 3 + j] - logf(-logf(gu[e * 3 + j]));
    if (z > best) { best = z; bi = j; }
  }
  sidx[e] = bi;                              // sign = sidx - 1
}
// table[s][f] = semb[s] @ a1_w[2*VAL: , f]
__global__ void k_table3(const float* __restrict__ semb, const float* __restrict__ a1w,
                         float* __restrict__ table) {
  int t = blockIdx.x * blockDim.x + threadIdx.x;
  if (t >= 3 * DIM_HID) return;
  int s = t >> 7, f = t & 127;
  float acc = 0.0f;
#pragma unroll
  for (int j = 0; j < DIM_SEMB; ++j)
    acc += semb[s * DIM_SEMB + j] * a1w[(2 * DIM_VAL + j) * DIM_HID + f];
  table[t] = acc;
}
__global__ void k_attn_epi(float* __restrict__ ebuf, const float* __restrict__ table,
                           const float* __restrict__ a1b, const int* __restrict__ sidx) {
  int t = blockIdx.x * blockDim.x + threadIdx.x;
  if (t >= N_EDGES * DIM_HID) return;
  int e = t >> 7, f = t & 127;
  ebuf[t] = fmaxf(ebuf[t] + table[sidx[e] * DIM_HID + f] + a1b[f], 0.0f);
}
// alpha -> softshrink -> per-edge coefficient (+|alpha| reduction for last layer)
__global__ void k_alpha(const float* __restrict__ ebuf, const float* __restrict__ a2w,
                        const float* __restrict__ a2b, const int* __restrict__ sidx,
                        float* __restrict__ coef, float* __restrict__ scal, int doAbs) {
  int e = blockIdx.x * blockDim.x + threadIdx.x;
  if (e >= N_EDGES) return;
  const float* hr = ebuf + (long long)e * DIM_HID;
  float a = a2b[0];
  for (int k2 = 0; k2 < DIM_HID; ++k2) a += hr[k2] * a2w[k2];
  a = a > LAM_F ? a - LAM_F : (a < -LAM_F ? a + LAM_F : 0.0f);
  int sg = sidx[e] - 1;
  coef[e] = sg > 0 ? a : (sg < 0 ? -GAMMA_F * fabsf(a) : 0.0f);
  if (doAbs) atomAddF(&scal[0], fabsf(a));
}
// out[col] += coef[e] * V[row], 4 feats/thread
__global__ void k_sagg(const float* __restrict__ V, const int* __restrict__ ei,
                       const float* __restrict__ coef, float* __restrict__ out) {
  int t = blockIdx.x * blockDim.x + threadIdx.x;
  if (t >= N_EDGES * 32) return;
  int e = t >> 5, f = (t & 31) << 2;
  float cf = coef[e];
  if (cf == 0.0f) return;
  int r = ei[e], c = ei[N_EDGES + e];
  const float4 v = *(const float4*)(V + r * DIM_VAL + f);
  float* o = out + c * DIM_HID + f;
  atomAddF(o + 0, cf * v.x); atomAddF(o + 1, cf * v.y);
  atomAddF(o + 2, cf * v.z); atomAddF(o + 3, cf * v.w);
}
__global__ void k_cls_acc(const float* __restrict__ logits, float* __restrict__ pacc) {
  int i = blockIdx.x * blockDim.x + threadIdx.x;
  if (i >= N_NODES) return;
  const float* lr = logits + i * N_CLS;
  float m = -1e30f;
#pragma unroll
  for (int j = 0; j < N_CLS; ++j) m = fmaxf(m, lr[j]);
  float p[N_CLS], s = 0.0f;
#pragma unroll
  for (int j = 0; j < N_CLS; ++j) { p[j] = expf(lr[j] - m); s += p[j]; }
  float inv = 1.0f / s;
#pragma unroll
  for (int j = 0; j < N_CLS; ++j) pacc[i * N_CLS + j] += p[j] * inv;
}
__global__ void k_finalize(const float* __restrict__ pacc, const float* __restrict__ scal,
                           float* __restrict__ out) {
  int i = blockIdx.x * blockDim.x + threadIdx.x;
  const int NC = N_NODES * N_CLS;
  if (i < NC) out[i] = logf(pacc[i] * (1.0f / K_MC) + 1e-12f);
  if (i == 0) {
    out[NC]     = scal[0] / (float)(N_EDGES * K_MC);          // sparse_loss
    out[NC + 1] = (scal[1] - scal[2]) / (float)N_EDGES;       // struct_loss
  }
}
// W[Kd rows starting at rowOff][Nd] (f32, row-major) -> BT[Nd][Kd] bf16
__global__ void k_wT(const float* __restrict__ W, __bf16* __restrict__ BT,
                     int Kdim, int Ndim, int rowOff) {
  int t = blockIdx.x * blockDim.x + threadIdx.x;
  if (t >= Kdim * Ndim) return;
  int n = t / Kdim, k2 = t % Kdim;
  BT[(long long)n * Kdim + k2] = f2bf(W[(long long)(rowOff + k2) * Ndim + n]);
}

// ---------------------------------------------------------------------------
extern "C" void kernel_launch(void* const* d_in, const int* in_sizes, int n_in,
                              void* d_out, int out_size, void* d_ws, size_t ws_size,
                              hipStream_t stream)
{
  (void)in_sizes; (void)n_in; (void)out_size; (void)ws_size;
  // inputs (setup_inputs dict order, params in insertion order)
  const float* x     = (const float*)d_in[0];
  const int*   ei    = (const int*)  d_in[1];   // [2,E]: row then col
  const float* gum   = (const float*)d_in[2];   // [K,E,3]
  const float* bb1_w = (const float*)d_in[3];
  const float* bb1_b = (const float*)d_in[4];
  const float* bb2_w = (const float*)d_in[5];
  const float* bb2_b = (const float*)d_in[6];
  const float* bbp_w = (const float*)d_in[7];
  const float* se1_w = (const float*)d_in[8];
  const float* se1_b = (const float*)d_in[9];
  const float* se2_w = (const float*)d_in[10];
  const float* se2_b = (const float*)d_in[11];
  const float* em1_w = (const float*)d_in[12];
  const float* em1_b = (const float*)d_in[13];
  const float* em2_w = (const float*)d_in[14];
  const float* em2_b = (const float*)d_in[15];
  const float* cls_w = (const float*)d_in[16];
  const float* cls_b = (const float*)d_in[17];
  const float *L_wv[2], *L_wt[2], *L_semb[2], *L_a1w[2], *L_a1b[2],
              *L_a2w[2], *L_a2b[2], *L_wself[2], *L_wout[2], *L_woutb[2];
  for (int l = 0; l < 2; ++l) {
    int b = 18 + l * 10;
    L_wv[l]    = (const float*)d_in[b + 0];
    L_wt[l]    = (const float*)d_in[b + 1];
    L_semb[l]  = (const float*)d_in[b + 2];
    L_a1w[l]   = (const float*)d_in[b + 3];
    L_a1b[l]   = (const float*)d_in[b + 4];
    L_a2w[l]   = (const float*)d_in[b + 5];
    L_a2b[l]   = (const float*)d_in[b + 6];
    L_wself[l] = (const float*)d_in[b + 7];
    L_wout[l]  = (const float*)d_in[b + 8];
    L_woutb[l] = (const float*)d_in[b + 9];
  }
  const int* rowp = ei;
  const int* colp = ei + N_EDGES;

  // ---- workspace bump allocator (256B aligned; ~250 MB total) ----
  char* wsp = (char*)d_ws; size_t off = 0;
  auto alloc = [&](size_t bytes) -> void* {
    void* p = wsp + off; off += (bytes + 255) & ~(size_t)255; return p;
  };
  const int NH = N_NODES * DIM_HID;
  const int EH = N_EDGES * DIM_HID;
  float* dinv = (float*)alloc(N_NODES * 4);
  float* t0   = (float*)alloc((size_t)NH * 4);
  float* aggb = (float*)alloc((size_t)NH * 4);
  float* h1   = (float*)alloc((size_t)NH * 4);
  float* xp   = (float*)alloc((size_t)NH * 4);
  float* H0   = (float*)alloc((size_t)NH * 4);
  float* hs1  = (float*)alloc((size_t)NH * 4);
  float* hs   = (float*)alloc((size_t)NH * 4);
  float* Vb   = (float*)alloc((size_t)NH * 4);
  float* Tb   = (float*)alloc((size_t)NH * 4);
  float* t1   = (float*)alloc((size_t)NH * 4);
  float* HA   = (float*)alloc((size_t)NH * 4);
  float* HB   = (float*)alloc((size_t)NH * 4);
  float* ebuf = (float*)alloc((size_t)EH * 4);
  float* elog = (float*)alloc((size_t)N_EDGES * 3 * 4);
  int*   sidx = (int*)  alloc((size_t)N_EDGES * 4);
  float* coef = (float*)alloc((size_t)N_EDGES * 4);
  float* clsb = (float*)alloc((size_t)N_NODES * N_CLS * 4);
  float* pacc = (float*)alloc((size_t)N_NODES * N_CLS * 4);
  float* tab[2] = { (float*)alloc(3 * DIM_HID * 4), (float*)alloc(3 * DIM_HID * 4) };
  float* scal = (float*)alloc(4 * 4);
  // bf16 activation shadows (A operands of all WMMA GEMMs)
  auto abf = [&](size_t n) { return (__bf16*)alloc(n * 2); };
  __bf16* xb   = abf((size_t)N_NODES * DIM_IN);
  __bf16* h1b  = abf(NH);  __bf16* hs1b = abf(NH); __bf16* hsb  = abf(NH);
  __bf16* H0b  = abf(NH);  __bf16* HAb  = abf(NH); __bf16* HBb  = abf(NH);
  __bf16* Vbb  = abf(NH);  __bf16* Tbb  = abf(NH); __bf16* aggbb = abf(NH);
  // bf16 transposed weights
  __bf16 *bb1T = abf(256 * 128), *bb2T = abf(128 * 128), *bbpT = abf(256 * 128),
         *se1T = abf(256 * 128), *se2T = abf(128 * 128),
         *em1aT = abf(128 * 128), *em1bT = abf(128 * 128), *clsT = abf(128 * 16);
  __bf16 *wvT[2], *wtT[2], *a1tT[2], *a1vT[2], *wselfT[2], *woutT[2];
  for (int l = 0; l < 2; ++l) {
    wvT[l] = abf(128 * 128); wtT[l] = abf(128 * 128);
    a1tT[l] = abf(128 * 128); a1vT[l] = abf(128 * 128);
    wselfT[l] = abf(128 * 128); woutT[l] = abf(128 * 128);
  }

  const int TB = 256;
  auto blocks = [](long long n) { return (unsigned)((n + 255) / 256); };
  auto gemm = [&](const __bf16* A, const int* g, const __bf16* BT,
                  const float* bias, const float* resid, float* Cm, __bf16* Cbf,
                  int M, int Kd, int Nd, int flags) {
    if (Nd % 32 == 0) {
      dim3 gr((unsigned)(M / 16), (unsigned)(Nd / 32));
      gemm_wmma<2><<<gr, 32, 0, stream>>>(A, g, BT, bias, resid, Cm, Cbf, Kd, Nd, flags);
    } else {
      dim3 gr((unsigned)(M / 16), (unsigned)(Nd / 16));
      gemm_wmma<1><<<gr, 32, 0, stream>>>(A, g, BT, bias, resid, Cm, Cbf, Kd, Nd, flags);
    }
  };
  auto conv = [&](const float* W, __bf16* BT, int Kd, int Nd, int roff) {
    k_wT<<<blocks((long long)Kd * Nd), TB, 0, stream>>>(W, BT, Kd, Nd, roff);
  };

  // ---- per-call init, degrees, weight conversion ----
  k_fill<<<blocks(4), TB, 0, stream>>>(scal, 4, 0.0f);
  k_fill<<<blocks(N_NODES * N_CLS), TB, 0, stream>>>(pacc, N_NODES * N_CLS, 0.0f);
  k_fill<<<blocks(N_NODES), TB, 0, stream>>>(dinv, N_NODES, 1.0f);   // self loop
  k_deg<<<blocks(N_EDGES), TB, 0, stream>>>(ei, dinv);
  k_rsqrt<<<blocks(N_NODES), TB, 0, stream>>>(dinv, N_NODES);
  k_f2bf16<<<blocks((long long)N_NODES * DIM_IN), TB, 0, stream>>>(x, xb, N_NODES * DIM_IN);
  conv(bb1_w, bb1T, 256, 128, 0);  conv(bb2_w, bb2T, 128, 128, 0);
  conv(bbp_w, bbpT, 256, 128, 0);  conv(se1_w, se1T, 256, 128, 0);
  conv(se2_w, se2T, 128, 128, 0);
  conv(em1_w, em1aT, 128, 128, 0); conv(em1_w, em1bT, 128, 128, 128);
  conv(cls_w, clsT, 128, 16, 0);
  for (int l = 0; l < 2; ++l) {
    conv(L_wv[l], wvT[l], 128, 128, 0);    conv(L_wt[l], wtT[l], 128, 128, 0);
    conv(L_a1w[l], a1tT[l], 128, 128, 0);  conv(L_a1w[l], a1vT[l], 128, 128, 128);
    conv(L_wself[l], wselfT[l], 128, 128, 0); conv(L_wout[l], woutT[l], 128, 128, 0);
    k_table3<<<blocks(3 * DIM_HID), TB, 0, stream>>>(L_semb[l], L_a1w[l], tab[l]);
  }

  // ---- GCN backbone ----
  gemm(xb, nullptr, bb1T, nullptr, nullptr, t0, nullptr, N_NODES, 256, 128, 0);
  k_selfscale<<<blocks(NH), TB, 0, stream>>>(t0, dinv, aggb);
  k_gcn_edge<<<blocks((long long)N_EDGES * 32), TB, 0, stream>>>(t0, ei, dinv, aggb);
  k_epilogue<<<blocks(NH), TB, 0, stream>>>(aggb, bb1_b, nullptr, h1, h1b, NH, 1 | 4 | 8);

  gemm(h1b, nullptr, bb2T, nullptr, nullptr, t0, nullptr, N_NODES, 128, 128, 0);
  k_selfscale<<<blocks(NH), TB, 0, stream>>>(t0, dinv, aggb);
  k_gcn_edge<<<blocks((long long)N_EDGES * 32), TB, 0, stream>>>(t0, ei, dinv, aggb);
  gemm(xb, nullptr, bbpT, nullptr, nullptr, xp, nullptr, N_NODES, 256, 128, 0);
  k_epilogue<<<blocks(NH), TB, 0, stream>>>(aggb, bb2_b, xp, H0, H0b, NH, 1 | 2 | 4 | 8);

  // ---- structural encoder ----
  gemm(xb, nullptr, se1T, nullptr, nullptr, t0, nullptr, N_NODES, 256, 128, 0);
  k_selfscale<<<blocks(NH), TB, 0, stream>>>(t0, dinv, aggb);
  k_gcn_edge<<<blocks((long long)N_EDGES * 32), TB, 0, stream>>>(t0, ei, dinv, aggb);
  k_epilogue<<<blocks(NH), TB, 0, stream>>>(aggb, se1_b, nullptr, hs1, hs1b, NH, 1 | 4 | 8);

  gemm(hs1b, nullptr, se2T, nullptr, nullptr, t0, nullptr, N_NODES, 128, 128, 0);
  k_selfscale<<<blocks(NH), TB, 0, stream>>>(t0, dinv, aggb);
  k_gcn_edge<<<blocks((long long)N_EDGES * 32), TB, 0, stream>>>(t0, ei, dinv, aggb);
  k_epilogue<<<blocks(NH), TB, 0, stream>>>(aggb, se2_b, nullptr, hs, hsb, NH, 1 | 8);

  // ---- edge MLP -> logits, struct loss ----
  gemm(hsb, rowp, em1aT, nullptr, nullptr, ebuf, nullptr, N_EDGES, 128, 128, 0);
  gemm(hsb, colp, em1bT, nullptr, nullptr, ebuf, nullptr, N_EDGES, 128, 128, 1);
  k_epilogue<<<blocks(EH), TB, 0, stream>>>(ebuf, em1_b, nullptr, ebuf, nullptr, EH, 1 | 4);
  k_edge_logits<<<blocks(N_EDGES), TB, 0, stream>>>(ebuf, em2_w, em2_b, elog);
  k_struct_loss<<<blocks(N_EDGES), TB, 0, stream>>>(elog, scal);

  // ---- K Monte-Carlo samples ----
  for (int k = 0; k < K_MC; ++k) {
    k_sign<<<blocks(N_EDGES), TB, 0, stream>>>(elog, gum + (size_t)k * N_EDGES * 3, sidx);
    const __bf16* Hinb = H0b;
    const float*  Hin  = H0;
    float*  Houts[2]  = { HA, HB };
    __bf16* Houtsb[2] = { HAb, HBb };
    for (int l = 0; l < 2; ++l) {
      gemm(Hinb, nullptr, wvT[l], nullptr, nullptr, Vb, Vbb, N_NODES, 128, 128, 16);
      gemm(Hinb, nullptr, wtT[l], nullptr, nullptr, Tb, Tbb, N_NODES, 128, 128, 16);
      gemm(Tbb, colp, a1tT[l], nullptr, nullptr, ebuf, nullptr, N_EDGES, 128, 128, 0);
      gemm(Vbb, rowp, a1vT[l], nullptr, nullptr, ebuf, nullptr, N_EDGES, 128, 128, 1);
      k_attn_epi<<<blocks(EH), TB, 0, stream>>>(ebuf, tab[l], L_a1b[l], sidx);
      k_alpha<<<blocks(N_EDGES), TB, 0, stream>>>(ebuf, L_a2w[l], L_a2b[l], sidx,
                                                  coef, scal, (l == 1) ? 1 : 0);
      k_fill<<<blocks(NH), TB, 0, stream>>>(aggb, NH, 0.0f);
      k_sagg<<<blocks((long long)N_EDGES * 32), TB, 0, stream>>>(Vb, ei, coef, aggb);
      k_f2bf16<<<blocks(NH), TB, 0, stream>>>(aggb, aggbb, NH);
      gemm(aggbb, nullptr, woutT[l], nullptr, nullptr, t1, nullptr, N_NODES, 128, 128, 0);
      gemm(Hinb, nullptr, wselfT[l], nullptr, nullptr, t1, nullptr, N_NODES, 128, 128, 1);
      k_epilogue<<<blocks(NH), TB, 0, stream>>>(t1, L_woutb[l], Hin, Houts[l], Houtsb[l],
                                                NH, 1 | 2 | 4 | 8);
      Hin = Houts[l]; Hinb = Houtsb[l];
    }
    gemm(Hinb, nullptr, clsT, cls_b, nullptr, clsb, nullptr, N_NODES, 128, 16, 2);
    k_cls_acc<<<blocks(N_NODES), TB, 0, stream>>>(clsb, pacc);
  }

  k_finalize<<<blocks((long long)N_NODES * N_CLS), TB, 0, stream>>>(pacc, scal, (float*)d_out);
}
